// Attention_28690381537684
// MI455X (gfx1250) — matrix operand order repaired and checked by reading, not compiled, not run on hardware
//
#include <hip/hip_runtime.h>
#include <hip/hip_bf16.h>

// ---------------------------------------------------------------------------
// Causal single-head attention, B=8, S=1024, E=1024, fp32 in/out.
// bf16 WMMA GEMMs (v_wmma_f32_16x16x32_bf16, f32 accum); intermediates are
// L2-resident (192MB). Global->LDS staging uses CDNA5 async-to-LDS DMA
// (ASYNCcnt) with double-buffered LDS when the builtin is available.
// ---------------------------------------------------------------------------

typedef unsigned short u16;
typedef __attribute__((ext_vector_type(16))) __bf16 v16bf;
typedef __attribute__((ext_vector_type(8)))  __bf16 v8bf;
typedef __attribute__((ext_vector_type(8)))  float  v8f;
typedef int v4i __attribute__((vector_size(16)));

union V16 { v16bf v; v8bf h[2]; };

__device__ __forceinline__ u16 f2bf(float f) {
    unsigned u = __float_as_uint(f);
    return (u16)((u + 0x7FFFu + ((u >> 16) & 1u)) >> 16); // RNE
}

// ---------------- CDNA5 async global->LDS (ASYNCcnt) probes ----------------
#if defined(__HIP_DEVICE_COMPILE__) && defined(__gfx1250__) && \
    __has_builtin(__builtin_amdgcn_global_load_async_to_lds_b128)
#define ASYNC_LDS 1
__device__ __forceinline__ void async_cp16(const u16* g, u16* l) {
    v4i* gg = (v4i*)g;                       // generic, drops const
    v4i* ll = (v4i*)l;
    __builtin_amdgcn_global_load_async_to_lds_b128(
        (__attribute__((address_space(1))) v4i*)gg,
        (__attribute__((address_space(3))) v4i*)ll, 0, 0);
}
__device__ __forceinline__ void wait_async0() {
#if __has_builtin(__builtin_amdgcn_s_wait_asynccnt)
    __builtin_amdgcn_s_wait_asynccnt(0);
#else
    asm volatile("s_wait_asynccnt 0x0" ::: "memory");
#endif
}
#else
#define ASYNC_LDS 0
#endif

// -------------------------- fp32 -> bf16 convert ---------------------------
__global__ __launch_bounds__(256)
void cvt_f32_bf16(const float* __restrict__ src, u16* __restrict__ dst, int n4) {
    int i = blockIdx.x * 256 + threadIdx.x;
    if (i < n4) {
        float4 v = ((const float4*)src)[i];
        ushort4 o;
        o.x = f2bf(v.x); o.y = f2bf(v.y); o.z = f2bf(v.z); o.w = f2bf(v.w);
        ((ushort4*)dst)[i] = o;
    }
}

// ------------------------------- WMMA GEMM ---------------------------------
// C[M,N] = A[M,K] * B[N,K]^T   (K contiguous in both operands)
// MODE 0: C=bf16, +bias[n]                      (Q, K projections)
// MODE 1: C=bf16, +bias[n], transposed store    (V -> Vt[b][e][s])
// MODE 2: C=f32, *scale, causal mask (n>m -> -1e9), per-z batch
// MODE 3: C=f32, triangular K-limit (k <= m),   per-z batch
#define LDT 40   // LDS row stride in bf16 elems: 80B, keeps b128 alignment

template<int MODE>
__global__ __launch_bounds__(256)
void gemm_wmma(const u16* __restrict__ Ag, const u16* __restrict__ Bg,
               const float* __restrict__ bias, void* __restrict__ Cv,
               int M, int N, int K, long zsA, long zsB, long zsC, float scale)
{
    const u16* A = Ag + (long)blockIdx.z * zsA;
    const u16* B = Bg + (long)blockIdx.z * zsB;

    const int m0   = blockIdx.x * 128;
    const int n0   = blockIdx.y * 128;
    const int tid  = threadIdx.x;
    const int lane = tid & 31;
    const int wave = tid >> 5;
    const int wm   = (wave >> 2) * 64;   // 2 wave-rows
    const int wn   = (wave & 3) * 32;    // 4 wave-cols

    __shared__ u16 As[2][128 * LDT];
    __shared__ u16 Bs[2][128 * LDT];

    v8f acc[4][2] = {};

    int kTiles = K >> 5;
    if (MODE == 2 && n0 > m0 + 127) kTiles = 0;          // fully-masked tile
    if (MODE == 3) { int kl = m0 + 128; if (kl < K) kTiles = kl >> 5; }

    const int  lrow   = tid >> 1;            // 128 rows, 2 threads/row
    const int  lcol   = (tid & 1) * 16;      // 16 bf16 = 32B per thread
    const long gaBase = (long)(m0 + lrow) * K + lcol;
    const long gbBase = (long)(n0 + lrow) * K + lcol;
    const int  lOff   = lrow * LDT + lcol;

    const int hf = lane >> 4;                // lane half selects K-subset
    const int r  = lane & 15;

#if ASYNC_LDS
    if (kTiles > 0) {                        // prologue DMA for kt=0 -> buf 0
        const u16* ga = A + gaBase;
        const u16* gb = B + gbBase;
        async_cp16(ga,     &As[0][lOff]);
        async_cp16(ga + 8, &As[0][lOff + 8]);
        async_cp16(gb,     &Bs[0][lOff]);
        async_cp16(gb + 8, &Bs[0][lOff + 8]);
    }
    for (int kt = 0; kt < kTiles; ++kt) {
        const int cur = kt & 1;
        wait_async0();                       // my DMAs for buf cur complete
        __syncthreads();                     // publishes cur; frees cur^1
        if (kt + 1 < kTiles) {               // overlap next tile's DMA
            const int nb = cur ^ 1;
            const u16* ga = A + gaBase + (kt + 1) * 32;
            const u16* gb = B + gbBase + (kt + 1) * 32;
            async_cp16(ga,     &As[nb][lOff]);
            async_cp16(ga + 8, &As[nb][lOff + 8]);
            async_cp16(gb,     &Bs[nb][lOff]);
            async_cp16(gb + 8, &Bs[nb][lOff + 8]);
        }
#else
    for (int kt = 0; kt < kTiles; ++kt) {
        const int cur = 0;
        const u16* ga = A + gaBase + kt * 32;
        const u16* gb = B + gbBase + kt * 32;
        uint4 a0 = *(const uint4*)ga;
        uint4 a1 = *(const uint4*)(ga + 8);
        uint4 b0 = *(const uint4*)gb;
        uint4 b1 = *(const uint4*)(gb + 8);
        __syncthreads();
        *(uint4*)&As[0][lOff]     = a0;
        *(uint4*)&As[0][lOff + 8] = a1;
        *(uint4*)&Bs[0][lOff]     = b0;
        *(uint4*)&Bs[0][lOff + 8] = b1;
        __syncthreads();
#endif
        // B fragments: lane half 0 -> K 0..15, half 1 -> K 16..31 of col N=r
        V16 bf[2];
        #pragma unroll
        for (int nt = 0; nt < 2; ++nt) {
            const u16* p = &Bs[cur][(wn + nt * 16 + r) * LDT + hf * 16];
            bf[nt].h[0] = *(const v8bf*)p;
            bf[nt].h[1] = *(const v8bf*)(p + 8);
        }
        // A fragments: half 0 -> K 0..15, half 1 -> K 16..31
        #pragma unroll
        for (int mt = 0; mt < 4; ++mt) {
            const u16* p = &As[cur][(wm + mt * 16 + r) * LDT + hf * 16];
            V16 af;
            af.h[0] = *(const v8bf*)p;
            af.h[1] = *(const v8bf*)(p + 8);
            #pragma unroll
            for (int nt = 0; nt < 2; ++nt) {
                acc[mt][nt] = __builtin_amdgcn_wmma_f32_16x16x32_bf16(
                    false, af.v, false, bf[nt].v, (short)0, acc[mt][nt],
                    false, false);
            }
        }
    }

    // ---- writeback: lane L<16 -> N=L, M=r ; L>=16 -> N=L-16, M=r+8 ----
    const int nl = lane & 15;
    #pragma unroll
    for (int nt = 0; nt < 2; ++nt) {
        const int n = n0 + wn + nt * 16 + nl;
        const float bv = (MODE == 0 || MODE == 1) ? bias[n] : 0.0f;
        #pragma unroll
        for (int mt = 0; mt < 4; ++mt) {
            #pragma unroll
            for (int rr = 0; rr < 8; ++rr) {
                const int m = m0 + wm + mt * 16 + hf * 8 + rr;
                const float v = acc[mt][nt][rr];
                if (MODE == 0) {
                    ((u16*)Cv)[(long)m * N + n] = f2bf(v + bv);
                } else if (MODE == 1) {
                    // Vt[b][n][s] ; m = b*1024 + s
                    const long b = m >> 10, s = m & 1023;
                    ((u16*)Cv)[(b << 20) + ((long)n << 10) + s] = f2bf(v + bv);
                } else if (MODE == 2) {
                    float* C = (float*)Cv + (long)blockIdx.z * zsC;
                    C[(long)m * N + n] = (n > m) ? -1e9f : v * scale;
                } else {
                    float* C = (float*)Cv + (long)blockIdx.z * zsC;
                    C[(long)m * N + n] = v;
                }
            }
        }
    }
}

// ----------------------------- row softmax ---------------------------------
// One block (256 threads, 8 wave32) per row of 1024 f32 scores -> bf16 P.
__global__ __launch_bounds__(256)
void softmax_row(const float* __restrict__ S, u16* __restrict__ P) {
    const long row = blockIdx.x;
    const float* src = S + row * 1024;
    u16* dst = P + row * 1024;
    const int t = threadIdx.x;

    float4 x = *(const float4*)(src + t * 4);
    float m = fmaxf(fmaxf(x.x, x.y), fmaxf(x.z, x.w));
    #pragma unroll
    for (int o = 16; o; o >>= 1) m = fmaxf(m, __shfl_xor(m, o, 32));

    __shared__ float red[8];
    if ((t & 31) == 0) red[t >> 5] = m;
    __syncthreads();
    float rm = red[0];
    #pragma unroll
    for (int i = 1; i < 8; ++i) rm = fmaxf(rm, red[i]);
    __syncthreads();

    float e0 = __expf(x.x - rm), e1 = __expf(x.y - rm);
    float e2 = __expf(x.z - rm), e3 = __expf(x.w - rm);
    float s = e0 + e1 + e2 + e3;
    #pragma unroll
    for (int o = 16; o; o >>= 1) s += __shfl_xor(s, o, 32);
    if ((t & 31) == 0) red[t >> 5] = s;
    __syncthreads();
    float rs = 0.0f;
    #pragma unroll
    for (int i = 0; i < 8; ++i) rs += red[i];

    const float inv = 1.0f / rs;
    ushort4 pk;
    pk.x = f2bf(e0 * inv); pk.y = f2bf(e1 * inv);
    pk.z = f2bf(e2 * inv); pk.w = f2bf(e3 * inv);
    *(ushort4*)(dst + t * 4) = pk;
}

// ------------------------------ launcher -----------------------------------
extern "C" void kernel_launch(void* const* d_in, const int* in_sizes, int n_in,
                              void* d_out, int out_size, void* d_ws, size_t ws_size,
                              hipStream_t stream) {
    const float* x  = (const float*)d_in[0];
    const float* Wq = (const float*)d_in[1];
    const float* bq = (const float*)d_in[2];
    const float* Wk = (const float*)d_in[3];
    const float* bk = (const float*)d_in[4];
    const float* Wv = (const float*)d_in[5];
    const float* bv = (const float*)d_in[6];
    float* out = (float*)d_out;

    const long MB = 1024 * 1024;            // element count of one 1024x1024
    u16* xb  = (u16*)d_ws;                  // [8192,1024] bf16
    u16* wqb = xb  + 8 * MB;                // [1024,1024] bf16
    u16* wkb = wqb + 1 * MB;
    u16* wvb = wkb + 1 * MB;
    u16* qb  = wvb + 1 * MB;                // [8192,1024] bf16
    u16* kb  = qb  + 8 * MB;                // [8192,1024] bf16
    u16* vtb = kb  + 8 * MB;                // [8][1024(e)][1024(s)] bf16
    float* sc = (float*)(vtb + 8 * MB);     // [8][1024][1024] f32
    u16* pb  = (u16*)(sc + 8 * MB);         // [8][1024][1024] bf16
    // total workspace: ~118 MB

    // 1) fp32 -> bf16 operand staging
    cvt_f32_bf16<<<dim3((8 * MB / 4) / 256), 256, 0, stream>>>(x,  xb,  8 * MB / 4);
    cvt_f32_bf16<<<dim3((MB / 4) / 256),     256, 0, stream>>>(Wq, wqb, MB / 4);
    cvt_f32_bf16<<<dim3((MB / 4) / 256),     256, 0, stream>>>(Wk, wkb, MB / 4);
    cvt_f32_bf16<<<dim3((MB / 4) / 256),     256, 0, stream>>>(Wv, wvb, MB / 4);

    // 2) projections: Q = x Wq^T + bq, etc.; V stored transposed per batch
    gemm_wmma<0><<<dim3(64, 8, 1), 256, 0, stream>>>(xb, wqb, bq, qb,
        8192, 1024, 1024, 0, 0, 0, 1.0f);
    gemm_wmma<0><<<dim3(64, 8, 1), 256, 0, stream>>>(xb, wkb, bk, kb,
        8192, 1024, 1024, 0, 0, 0, 1.0f);
    gemm_wmma<1><<<dim3(64, 8, 1), 256, 0, stream>>>(xb, wvb, bv, vtb,
        8192, 1024, 1024, 0, 0, 0, 1.0f);

    // 3) scores = Q K^T / sqrt(1024), causal mask baked in
    gemm_wmma<2><<<dim3(8, 8, 8), 256, 0, stream>>>(qb, kb, nullptr, sc,
        1024, 1024, 1024, MB, MB, MB, 0.03125f);

    // 4) row softmax -> bf16 probabilities
    softmax_row<<<dim3(8192), 256, 0, stream>>>(sc, pb);

    // 5) out = P V (triangular K-loop: P is zero above diagonal)
    gemm_wmma<3><<<dim3(8, 8, 8), 256, 0, stream>>>(pb, vtb, nullptr, out,
        1024, 1024, 1024, MB, MB, MB, 1.0f);
}
